// LongformerAttention_3040836845917
// MI455X (gfx1250) — compile-verified, matrix operand order checked by reading, not run on hardware
//
#include <hip/hip_runtime.h>
#include <hip/hip_bf16.h>

typedef __bf16 bf16;
typedef __attribute__((ext_vector_type(16))) __bf16 v16bf;
typedef __attribute__((ext_vector_type(8)))  float  v8f;
typedef __attribute__((ext_vector_type(4)))  float  f32x4;

#define HIDDEN 1024
#define NHEAD  16
#define HDIM   64
#define WIN    64
#define SROW   200   // score row stride in floats (192 + pad)

// ---- WMMA helpers -----------------------------------------------------------

__device__ __forceinline__ v8f wmma_bf16(v16bf a, v16bf b, v8f c) {
  // D = A(16x32) * B(32x16) + C, fp32 accumulate
  return __builtin_amdgcn_wmma_f32_16x16x32_bf16(
      /*neg_a=*/false, a, /*neg_b=*/false, b,
      /*c_mod=*/(short)0, c, /*reuse_a=*/false, /*reuse_b=*/false);
}

// Load one 16-lane fragment slice: 16 contiguous bf16 split as [0..7] and [16..23]
// relative to p (p already includes the kb = (lane>>4)*8 offset). 16B aligned.
__device__ __forceinline__ v16bf load16(const bf16* p) {
  union { f32x4 q[2]; v16bf v; } u;
  u.q[0] = *(const f32x4*)(p);
  u.q[1] = *(const f32x4*)(p + 16);
  return u.v;
}

// ---- Prep kernels -----------------------------------------------------------

// X (fp32 [B,S0,1024]) -> Xp (bf16 [B,Sp,1024]), zero padded rows
__global__ void lf_convert_x(const float* __restrict__ X, bf16* __restrict__ Xp,
                             int S0, int Sp) {
  long i = (long)blockIdx.x * blockDim.x + threadIdx.x;
  long total = (long)2 * Sp * HIDDEN;
  if (i >= total) return;
  int col = (int)(i % HIDDEN);
  long row = i / HIDDEN;
  int b = (int)(row / Sp);
  int s = (int)(row % Sp);
  float v = (s < S0) ? X[((long)b * S0 + s) * HIDDEN + col] : 0.f;
  Xp[i] = (bf16)v;
}

// W [K,N] fp32 -> WT [N,K] bf16, 4 matrices selected by blockIdx.y
__global__ void lf_transpose_w(const float* __restrict__ w0, const float* __restrict__ w1,
                               const float* __restrict__ w2, const float* __restrict__ w3,
                               bf16* __restrict__ t0, bf16* __restrict__ t1,
                               bf16* __restrict__ t2, bf16* __restrict__ t3) {
  int z = blockIdx.y;
  const float* src = (z == 0) ? w0 : (z == 1) ? w1 : (z == 2) ? w2 : w3;
  bf16* dst = (z == 0) ? t0 : (z == 1) ? t1 : (z == 2) ? t2 : t3;
  long i = (long)blockIdx.x * blockDim.x + threadIdx.x;   // i = n*1024 + k
  int k = (int)(i & (HIDDEN - 1));
  int n = (int)(i >> 10);
  dst[i] = (bf16)src[(long)k * HIDDEN + n];
}

// ---- QKV projection GEMM ----------------------------------------------------
// C[M=B*Sp, N=1024] = Xp * W + bias ; z selects {Q,K,V}; V stored transposed.
// Block tile 128x128, wave tile 32x64, double-buffered fragment loads.
__global__ void lf_proj_gemm(const bf16* __restrict__ Xp,
                             const bf16* __restrict__ WTq, const bf16* __restrict__ WTk,
                             const bf16* __restrict__ WTv,
                             const float* __restrict__ bq, const float* __restrict__ bk,
                             const float* __restrict__ bv,
                             bf16* __restrict__ Q, bf16* __restrict__ K,
                             bf16* __restrict__ Vt,
                             int Sp, int Vstride) {
  int z = blockIdx.z;
  const bf16* WT = (z == 0) ? WTq : (z == 1) ? WTk : WTv;
  const float* bias = (z == 0) ? bq : (z == 1) ? bk : bv;
  bf16* out = (z == 0) ? Q : (z == 1) ? K : Vt;

  int wave = threadIdx.x >> 5;
  int lane = threadIdx.x & 31;
  int r  = lane & 15;
  int hi = lane >> 4;
  int kb = hi * 8;

  int mt2 = wave >> 1;                 // 0..3 -> 32-row strip
  int nt2 = wave & 1;                  // 0..1 -> 64-col strip
  int mbase = blockIdx.y * 128 + mt2 * 32;
  int nbase = blockIdx.x * 128 + nt2 * 64;

  const bf16* pa0 = Xp + (long)(mbase + r) * HIDDEN + kb;
  const bf16* pa1 = pa0 + (long)16 * HIDDEN;
  const bf16* pb0 = WT + (long)(nbase + 0 * 16 + r) * HIDDEN + kb;
  const bf16* pb1 = WT + (long)(nbase + 1 * 16 + r) * HIDDEN + kb;
  const bf16* pb2 = WT + (long)(nbase + 2 * 16 + r) * HIDDEN + kb;
  const bf16* pb3 = WT + (long)(nbase + 3 * 16 + r) * HIDDEN + kb;

  v8f acc[2][4] = {{v8f{}, v8f{}, v8f{}, v8f{}}, {v8f{}, v8f{}, v8f{}, v8f{}}};

  v16bf ca0 = load16(pa0), ca1 = load16(pa1);
  v16bf cb0 = load16(pb0), cb1 = load16(pb1), cb2 = load16(pb2), cb3 = load16(pb3);

#pragma unroll
  for (int kk = 0; kk < HIDDEN / 32; ++kk) {
    v16bf na0, na1, nb0, nb1, nb2, nb3;
    const bool more = (kk + 1) < (HIDDEN / 32);
    if (more) {
      int ko = (kk + 1) * 32;
      na0 = load16(pa0 + ko);  na1 = load16(pa1 + ko);
      nb0 = load16(pb0 + ko);  nb1 = load16(pb1 + ko);
      nb2 = load16(pb2 + ko);  nb3 = load16(pb3 + ko);
    }
    if ((kk + 4) < (HIDDEN / 32)) {      // prefetch ~4 k-steps ahead
      int kp = (kk + 4) * 32;
      __builtin_prefetch(pa0 + kp, 0, 0);
      __builtin_prefetch(pb0 + kp, 0, 0);
      __builtin_prefetch(pb2 + kp, 0, 0);
    }
    acc[0][0] = wmma_bf16(ca0, cb0, acc[0][0]);
    acc[1][0] = wmma_bf16(ca1, cb0, acc[1][0]);
    acc[0][1] = wmma_bf16(ca0, cb1, acc[0][1]);
    acc[1][1] = wmma_bf16(ca1, cb1, acc[1][1]);
    acc[0][2] = wmma_bf16(ca0, cb2, acc[0][2]);
    acc[1][2] = wmma_bf16(ca1, cb2, acc[1][2]);
    acc[0][3] = wmma_bf16(ca0, cb3, acc[0][3]);
    acc[1][3] = wmma_bf16(ca1, cb3, acc[1][3]);
    if (more) {
      ca0 = na0; ca1 = na1;
      cb0 = nb0; cb1 = nb1; cb2 = nb2; cb3 = nb3;
    }
  }

#pragma unroll
  for (int ms = 0; ms < 2; ++ms) {
#pragma unroll
    for (int nt = 0; nt < 4; ++nt) {
      int n = nbase + nt * 16 + r;
      float bv_ = bias[n];
      int h = n >> 6;        // head
      int d = n & 63;        // dim within head
#pragma unroll
      for (int i = 0; i < 8; ++i) {
        int m = mbase + ms * 16 + i + hi * 8;
        int bb = m / Sp;
        int s  = m - bb * Sp;
        float v = acc[ms][nt][i] + bv_;
        long idx;
        if (z == 2) {  // V transposed: [B,H,D,Vstride], +64 guard offset
          idx = ((long)(bb * NHEAD + h) * HDIM + d) * Vstride + 64 + s;
        } else {       // Q,K: [B,H,Sp,D]
          idx = (((long)(bb * NHEAD + h) * Sp) + s) * HDIM + d;
        }
        out[idx] = (bf16)v;
      }
    }
  }
}

// ---- Banded attention: one block per (b, h, chunk) --------------------------
__global__ void lf_attn(const bf16* __restrict__ Q, const bf16* __restrict__ K,
                        const bf16* __restrict__ Vt, const float* __restrict__ mask,
                        bf16* __restrict__ ctx,
                        int Sp, int S0, int nC, int Vstride) {
  __shared__ __align__(16) bf16  sQ[64 * 64];
  __shared__ __align__(16) float sS[64 * SROW];

  int blk = blockIdx.x;
  int c = blk % nC;
  int h = (blk / nC) % NHEAD;
  int b = blk / (nC * NHEAD);

  const bf16* Qb = Q + ((long)(b * NHEAD + h) * Sp) * HDIM;
  const bf16* Kb = K + ((long)(b * NHEAD + h) * Sp) * HDIM;
  const bf16* Vb = Vt + ((long)(b * NHEAD + h) * HDIM) * Vstride;

  int tid = threadIdx.x;

  // stage Q chunk [64,64] into LDS
  for (int i = tid; i < 64 * 64; i += 256) {
    int r0 = i >> 6, d0 = i & 63;
    sQ[i] = Qb[(long)(c * WIN + r0) * HDIM + d0];
  }
  __syncthreads();

  int wave = tid >> 5;
  int lane = tid & 31;
  int r  = lane & 15;
  int hi = lane >> 4;
  int kb = hi * 8;

  // ---- pass 1: scores S[64,192] = Qc * Kwin^T (pipelined K loads) ----
  {
    int mt = wave >> 1;                   // 0..3
    const bf16* qrow = sQ + (mt * 16 + r) * HDIM + kb;
    v16bf a0 = load16(qrow);              // d = 0..31 slice
    v16bf a1 = load16(qrow + 32);         // d = 32..63 slice
    int ntbase = (wave & 1) * 6;

    auto krowp = [&](int j) -> const bf16* {
      int key = (ntbase + j) * 16 + r;
      int sk = c * WIN - WIN + key;
      sk = sk < 0 ? 0 : (sk >= Sp ? Sp - 1 : sk);  // clamp; masked later
      return Kb + (long)sk * HDIM + kb;
    };

    v16bf b0 = load16(krowp(0));
    v16bf b1 = load16(krowp(0) + 32);
#pragma unroll
    for (int j = 0; j < 6; ++j) {
      v16bf n0, n1;
      if (j < 5) {
        const bf16* np = krowp(j + 1);
        n0 = load16(np);
        n1 = load16(np + 32);
      }
      v8f acc = v8f{};
      acc = wmma_bf16(a0, b0, acc);
      acc = wmma_bf16(a1, b1, acc);
      int key = (ntbase + j) * 16 + r;
#pragma unroll
      for (int i = 0; i < 8; ++i)
        sS[(mt * 16 + i + hi * 8) * SROW + key] = acc[i];
      if (j < 5) { b0 = n0; b1 = n1; }
    }
  }
  __syncthreads();

  // ---- pass 2: band mask + softmax, one thread per query row ----
  if (tid < 64) {
    int qi = tid;
    float* row = sS + qi * SROW;
    float mx = -1e30f;
    for (int kj = 0; kj < 192; ++kj) {
      float v = row[kj] * 0.125f;         // 1/sqrt(64)
      int sk = c * WIN - WIN + kj;
      bool valid = (kj >= qi) && (kj <= qi + 2 * WIN) &&
                   (sk >= 0) && (sk < S0) && (mask[(long)b * S0 + sk] > 0.f);
      v = valid ? v : -1e9f;
      row[kj] = v;
      mx = fmaxf(mx, v);
    }
    float sum = 0.f;
    for (int kj = 0; kj < 192; ++kj) {
      float e = __expf(row[kj] - mx);
      row[kj] = e;
      sum += e;
    }
    float inv = 1.f / sum;
    for (int kj = 0; kj < 192; ++kj) row[kj] *= inv;
  }
  __syncthreads();

  // ---- pass 3: ctx[64,64] = P[64,192] * Vwin[192,64] (pipelined V loads) ----
  {
    int t0 = wave * 2;
    int mt = t0 >> 2;                     // 0..3 (both tiles of a wave share mt)
    v16bf afr[6];
#pragma unroll
    for (int ks = 0; ks < 6; ++ks) {
      const float* prow = sS + (mt * 16 + r) * SROW + ks * 32 + kb;
      v16bf f;
#pragma unroll
      for (int j = 0; j < 8; ++j) {
        f[j]     = (bf16)prow[j];
        f[j + 8] = (bf16)prow[16 + j];
      }
      afr[ks] = f;
    }
#pragma unroll
    for (int tt = 0; tt < 2; ++tt) {
      int nt = (t0 + tt) & 3;
      int d  = nt * 16 + r;
      const bf16* vrow = Vb + (long)d * Vstride;
      // +64 guard makes base always in-bounds
      v16bf bcur = load16(vrow + (c * WIN + kb));
      v8f acc = v8f{};
#pragma unroll
      for (int ks = 0; ks < 6; ++ks) {
        v16bf bn;
        if (ks < 5) bn = load16(vrow + (c * WIN + (ks + 1) * 32 + kb));
        acc = wmma_bf16(afr[ks], bcur, acc);
        if (ks < 5) bcur = bn;
      }
#pragma unroll
      for (int i = 0; i < 8; ++i) {
        int s = c * WIN + mt * 16 + i + hi * 8;
        if (s < S0)
          ctx[((long)b * S0 + s) * HIDDEN + h * HDIM + d] = (bf16)acc[i];
      }
    }
  }
}

// ---- Output projection: out[M,1024] = ctx * Wo + bo (fp32 out) --------------
// Block tile 128x128, wave tile 32x64, double-buffered fragment loads.
__global__ void lf_oproj(const bf16* __restrict__ ctx, const bf16* __restrict__ WoT,
                         const float* __restrict__ bo, float* __restrict__ out,
                         int Mtot) {
  int wave = threadIdx.x >> 5;
  int lane = threadIdx.x & 31;
  int r  = lane & 15;
  int hi = lane >> 4;
  int kb = hi * 8;

  int mt2 = wave >> 1;
  int nt2 = wave & 1;
  int mbase = blockIdx.y * 128 + mt2 * 32;
  int nbase = blockIdx.x * 128 + nt2 * 64;

  int mr0 = mbase + r;        if (mr0 >= Mtot) mr0 = Mtot - 1;   // load clamp
  int mr1 = mbase + 16 + r;   if (mr1 >= Mtot) mr1 = Mtot - 1;

  const bf16* pa0 = ctx + (long)mr0 * HIDDEN + kb;
  const bf16* pa1 = ctx + (long)mr1 * HIDDEN + kb;
  const bf16* pb0 = WoT + (long)(nbase + 0 * 16 + r) * HIDDEN + kb;
  const bf16* pb1 = WoT + (long)(nbase + 1 * 16 + r) * HIDDEN + kb;
  const bf16* pb2 = WoT + (long)(nbase + 2 * 16 + r) * HIDDEN + kb;
  const bf16* pb3 = WoT + (long)(nbase + 3 * 16 + r) * HIDDEN + kb;

  v8f acc[2][4] = {{v8f{}, v8f{}, v8f{}, v8f{}}, {v8f{}, v8f{}, v8f{}, v8f{}}};

  v16bf ca0 = load16(pa0), ca1 = load16(pa1);
  v16bf cb0 = load16(pb0), cb1 = load16(pb1), cb2 = load16(pb2), cb3 = load16(pb3);

#pragma unroll
  for (int kk = 0; kk < HIDDEN / 32; ++kk) {
    v16bf na0, na1, nb0, nb1, nb2, nb3;
    const bool more = (kk + 1) < (HIDDEN / 32);
    if (more) {
      int ko = (kk + 1) * 32;
      na0 = load16(pa0 + ko);  na1 = load16(pa1 + ko);
      nb0 = load16(pb0 + ko);  nb1 = load16(pb1 + ko);
      nb2 = load16(pb2 + ko);  nb3 = load16(pb3 + ko);
    }
    if ((kk + 4) < (HIDDEN / 32)) {
      int kp = (kk + 4) * 32;
      __builtin_prefetch(pa0 + kp, 0, 0);
      __builtin_prefetch(pb0 + kp, 0, 0);
      __builtin_prefetch(pb2 + kp, 0, 0);
    }
    acc[0][0] = wmma_bf16(ca0, cb0, acc[0][0]);
    acc[1][0] = wmma_bf16(ca1, cb0, acc[1][0]);
    acc[0][1] = wmma_bf16(ca0, cb1, acc[0][1]);
    acc[1][1] = wmma_bf16(ca1, cb1, acc[1][1]);
    acc[0][2] = wmma_bf16(ca0, cb2, acc[0][2]);
    acc[1][2] = wmma_bf16(ca1, cb2, acc[1][2]);
    acc[0][3] = wmma_bf16(ca0, cb3, acc[0][3]);
    acc[1][3] = wmma_bf16(ca1, cb3, acc[1][3]);
    if (more) {
      ca0 = na0; ca1 = na1;
      cb0 = nb0; cb1 = nb1; cb2 = nb2; cb3 = nb3;
    }
  }

#pragma unroll
  for (int ms = 0; ms < 2; ++ms) {
#pragma unroll
    for (int nt = 0; nt < 4; ++nt) {
      int n = nbase + nt * 16 + r;
      float bias = bo[n];
#pragma unroll
      for (int i = 0; i < 8; ++i) {
        int m = mbase + ms * 16 + i + hi * 8;
        if (m < Mtot) out[(long)m * HIDDEN + n] = acc[ms][nt][i] + bias;
      }
    }
  }
}

// ---- Launch -----------------------------------------------------------------

extern "C" void kernel_launch(void* const* d_in, const int* in_sizes, int n_in,
                              void* d_out, int out_size, void* d_ws, size_t ws_size,
                              hipStream_t stream) {
  const float* X    = (const float*)d_in[0];
  const float* mask = (const float*)d_in[1];
  const float* Wq   = (const float*)d_in[2];
  const float* bq   = (const float*)d_in[3];
  const float* Wk   = (const float*)d_in[4];
  const float* bk   = (const float*)d_in[5];
  const float* Wv   = (const float*)d_in[6];
  const float* bv   = (const float*)d_in[7];
  const float* Wo   = (const float*)d_in[8];
  const float* bo   = (const float*)d_in[9];
  float* out = (float*)d_out;

  const int B  = 2;
  const int S0 = in_sizes[1] / B;                  // 4096
  const int pad = WIN - (S0 % WIN);                // matches reference (>=1)
  const int Sp = S0 + pad;                         // 4160
  const int nC = Sp / WIN;                         // 65
  const int Vstride = Sp + 128;                    // guard bands for window reads

  // workspace carve-out (all bf16)
  bf16* ws = (bf16*)d_ws;
  size_t o = 0;
  bf16* Xp  = ws + o; o += (size_t)B * Sp * HIDDEN;
  bf16* WTq = ws + o; o += (size_t)HIDDEN * HIDDEN;
  bf16* WTk = ws + o; o += (size_t)HIDDEN * HIDDEN;
  bf16* WTv = ws + o; o += (size_t)HIDDEN * HIDDEN;
  bf16* WTo = ws + o; o += (size_t)HIDDEN * HIDDEN;
  bf16* Qb  = ws + o; o += (size_t)B * NHEAD * Sp * HDIM;
  bf16* Kb  = ws + o; o += (size_t)B * NHEAD * Sp * HDIM;
  bf16* Vt  = ws + o; o += (size_t)B * NHEAD * HDIM * Vstride;
  bf16* ctx = ws + o; o += (size_t)B * S0 * HIDDEN;

  // zero V (guard bands must be finite; masked probs are exactly 0)
  hipMemsetAsync(Vt, 0, (size_t)B * NHEAD * HDIM * Vstride * sizeof(bf16), stream);

  // X -> bf16 padded
  {
    long total = (long)B * Sp * HIDDEN;
    int blocks = (int)((total + 255) / 256);
    lf_convert_x<<<blocks, 256, 0, stream>>>(X, Xp, S0, Sp);
  }
  // weights -> transposed bf16
  lf_transpose_w<<<dim3((HIDDEN * HIDDEN) / 256, 4), 256, 0, stream>>>(
      Wq, Wk, Wv, Wo, WTq, WTk, WTv, WTo);

  // QKV projections: block 128x128, grid (N/128, M/128, 3)
  lf_proj_gemm<<<dim3(HIDDEN / 128, (B * Sp) / 128, 3), 256, 0, stream>>>(
      Xp, WTq, WTk, WTv, bq, bk, bv, Qb, Kb, Vt, Sp, Vstride);

  // banded attention
  lf_attn<<<dim3(B * NHEAD * nC), 256, 0, stream>>>(
      Qb, Kb, Vt, mask, ctx, Sp, S0, nC, Vstride);

  // output projection
  int Mtot = B * S0;
  lf_oproj<<<dim3(HIDDEN / 128, (Mtot + 127) / 128), 256, 0, stream>>>(
      ctx, WTo, bo, out, Mtot);
}
